// CodingClassifier_24867860644037
// MI455X (gfx1250) — compile-verified, matrix operand order checked by reading, not run on hardware
//
#include <hip/hip_runtime.h>
#include <hip/hip_bf16.h>

typedef __attribute__((ext_vector_type(16))) __bf16 v16bf;
typedef __attribute__((ext_vector_type(8)))  __bf16 bf16x8;
typedef __attribute__((ext_vector_type(4)))  __bf16 bf16x4;
typedef __attribute__((ext_vector_type(8)))  float  v8f;

#define BM 128
#define BN 64
#define BK 32
#define LDT 48   // LDS row stride in bf16 elements: 96B rows -> 16B aligned b128 reads, spread banks

union AFrag { v16bf v; bf16x8 h[2]; };

__device__ __forceinline__ bf16x4 cvt_hi(const float4 f) {
    bf16x4 r;
    r[0] = (__bf16)f.x; r[1] = (__bf16)f.y; r[2] = (__bf16)f.z; r[3] = (__bf16)f.w;
    return r;
}
__device__ __forceinline__ bf16x4 cvt_lo(const float4 f, const bf16x4 h) {
    bf16x4 r;
    r[0] = (__bf16)(f.x - (float)h[0]);
    r[1] = (__bf16)(f.y - (float)h[1]);
    r[2] = (__bf16)(f.z - (float)h[2]);
    r[3] = (__bf16)(f.w - (float)h[3]);
    return r;
}

// One 256-thread block per row: sum of `cols` floats.
__global__ __launch_bounds__(256)
void rowsum_kernel(const float* __restrict__ X, float* __restrict__ S, int cols) {
    __shared__ float sm[256];
    const int row = blockIdx.x;
    const float* p = X + (size_t)row * cols;
    float s = 0.0f;
    for (int c = threadIdx.x; c < cols; c += 256) s += p[c];
    sm[threadIdx.x] = s;
    __syncthreads();
    for (int o = 128; o > 0; o >>= 1) {
        if (threadIdx.x < o) sm[threadIdx.x] += sm[threadIdx.x + o];
        __syncthreads();
    }
    if (threadIdx.x == 0) S[row] = sm[0];
}

// D[n][k] = (2 * sum_c A[n][c]*B[k][c] + C - osum[n] - csum[k]) / K
// A: [Nrows][C] f32, B (code_book): [K][C] f32.
// bf16 hi/lo split: each f32 GEMM term computed as two bf16 WMMAs into an f32 accumulator.
__global__ __launch_bounds__(256)
void hamming_gemm_kernel(const float* __restrict__ A,
                         const float* __restrict__ B,
                         const float* __restrict__ osum,
                         const float* __restrict__ csum,
                         float* __restrict__ D,
                         int Nrows, int K, int C) {
    __shared__ __align__(16) __bf16 sAhi[BM * LDT];
    __shared__ __align__(16) __bf16 sAlo[BM * LDT];
    __shared__ __align__(16) __bf16 sB  [BN * LDT];

    const int t    = threadIdx.x;
    const int lane = t & 31;
    const int wid  = t >> 5;     // 8 waves
    const int wm   = wid & 3;    // 0..3 over M
    const int wn   = wid >> 2;   // 0..1 over N
    const int lrow = lane & 15;  // row/col within 16x16 tile
    const int half = lane >> 4;  // K-half selector per 16-bit WMMA layout

    const int mBase = blockIdx.y * BM;
    const int nBase = blockIdx.x * BN;

    const v8f vzero = {0.f, 0.f, 0.f, 0.f, 0.f, 0.f, 0.f, 0.f};
    v8f acc[2][2];
    acc[0][0] = vzero; acc[0][1] = vzero; acc[1][0] = vzero; acc[1][1] = vzero;

    const int kSteps = (C + BK - 1) / BK;

    float4 aReg[4];  // 128x32 f32 tile staged by 256 threads: 4 float4 each
    float4 bReg[2];  //  64x32 f32 tile: 2 float4 each

    // ---- prologue: global load of k-step 0 (assumes C % 4 == 0 for float4 groups) ----
    {
        const int kBase = 0;
#pragma unroll
        for (int i = 0; i < 4; ++i) {
            int idx = t + i * 256;
            int r = idx >> 3, cv = idx & 7;
            int gr = mBase + r, gc = kBase + cv * 4;
            float4 v = {0.f, 0.f, 0.f, 0.f};
            if (gr < Nrows && gc + 3 < C) v = *(const float4*)(A + (size_t)gr * C + gc);
            aReg[i] = v;
        }
#pragma unroll
        for (int i = 0; i < 2; ++i) {
            int idx = t + i * 256;
            int r = idx >> 3, cv = idx & 7;
            int gr = nBase + r, gc = kBase + cv * 4;
            float4 v = {0.f, 0.f, 0.f, 0.f};
            if (gr < K && gc + 3 < C) v = *(const float4*)(B + (size_t)gr * C + gc);
            bReg[i] = v;
        }
    }

    for (int ks = 0; ks < kSteps; ++ks) {
        __syncthreads();  // previous compute done reading LDS

        // ---- store staged registers to LDS as bf16 hi/lo ----
#pragma unroll
        for (int i = 0; i < 4; ++i) {
            int idx = t + i * 256;
            int r = idx >> 3, cv = idx & 7;
            bf16x4 hi = cvt_hi(aReg[i]);
            bf16x4 lo = cvt_lo(aReg[i], hi);
            *(bf16x4*)&sAhi[r * LDT + cv * 4] = hi;
            *(bf16x4*)&sAlo[r * LDT + cv * 4] = lo;
        }
#pragma unroll
        for (int i = 0; i < 2; ++i) {
            int idx = t + i * 256;
            int r = idx >> 3, cv = idx & 7;
            *(bf16x4*)&sB[r * LDT + cv * 4] = cvt_hi(bReg[i]);  // 0/1 entries exact in bf16
        }
        __syncthreads();

        // ---- prefetch next k-step from global while WMMAs run ----
        if (ks + 1 < kSteps) {
            const int kBase = (ks + 1) * BK;
#pragma unroll
            for (int i = 0; i < 4; ++i) {
                int idx = t + i * 256;
                int r = idx >> 3, cv = idx & 7;
                int gr = mBase + r, gc = kBase + cv * 4;
                float4 v = {0.f, 0.f, 0.f, 0.f};
                if (gr < Nrows && gc + 3 < C) v = *(const float4*)(A + (size_t)gr * C + gc);
                aReg[i] = v;
            }
#pragma unroll
            for (int i = 0; i < 2; ++i) {
                int idx = t + i * 256;
                int r = idx >> 3, cv = idx & 7;
                int gr = nBase + r, gc = kBase + cv * 4;
                float4 v = {0.f, 0.f, 0.f, 0.f};
                if (gr < K && gc + 3 < C) v = *(const float4*)(B + (size_t)gr * C + gc);
                bReg[i] = v;
            }
        }

        // ---- B fragments (K x 16, per 16-bit WMMA layout; sB stored [class][c] == B^T) ----
        AFrag bf[2];
#pragma unroll
        for (int nt = 0; nt < 2; ++nt) {
            int base = (wn * 32 + nt * 16 + lrow) * LDT;
            bf[nt].h[0] = *(const bf16x8*)&sB[base + half * 8];
            bf[nt].h[1] = *(const bf16x8*)&sB[base + 16 + half * 8];
        }

        // ---- A fragments + 8 WMMAs (2 m-tiles x 2 n-tiles x hi/lo) ----
#pragma unroll
        for (int mt = 0; mt < 2; ++mt) {
            int base = (wm * 32 + mt * 16 + lrow) * LDT;
            AFrag ahi, alo;
            ahi.h[0] = *(const bf16x8*)&sAhi[base + half * 8];
            ahi.h[1] = *(const bf16x8*)&sAhi[base + 16 + half * 8];
            alo.h[0] = *(const bf16x8*)&sAlo[base + half * 8];
            alo.h[1] = *(const bf16x8*)&sAlo[base + 16 + half * 8];
#pragma unroll
            for (int nt = 0; nt < 2; ++nt) {
                acc[mt][nt] = __builtin_amdgcn_wmma_f32_16x16x32_bf16(
                    false, ahi.v, false, bf[nt].v, (short)0, acc[mt][nt], false, false);
                acc[mt][nt] = __builtin_amdgcn_wmma_f32_16x16x32_bf16(
                    false, alo.v, false, bf[nt].v, (short)0, acc[mt][nt], false, false);
            }
        }
    }

    // ---- epilogue: D layout lane L -> n = L%16, m = 8*(L/16) + vgpr_index ----
    const float invK = 1.0f / (float)K;
    const float Cf   = (float)C;
#pragma unroll
    for (int mt = 0; mt < 2; ++mt) {
        const int rowBase = mBase + wm * 32 + mt * 16 + half * 8;
        float osv[8];
#pragma unroll
        for (int e = 0; e < 8; ++e) {
            int row = rowBase + e;
            osv[e] = (row < Nrows) ? osum[row] : 0.0f;
        }
#pragma unroll
        for (int nt = 0; nt < 2; ++nt) {
            const int col = nBase + wn * 32 + nt * 16 + lrow;
            if (col < K) {
                const float cs = csum[col];
#pragma unroll
                for (int e = 0; e < 8; ++e) {
                    int row = rowBase + e;
                    if (row < Nrows) {
                        D[(size_t)row * K + col] =
                            (2.0f * acc[mt][nt][e] + Cf - osv[e] - cs) * invK;
                    }
                }
            }
        }
    }
}

extern "C" void kernel_launch(void* const* d_in, const int* in_sizes, int n_in,
                              void* d_out, int out_size, void* d_ws, size_t ws_size,
                              hipStream_t stream) {
    const float* out_feats = (const float*)d_in[0];  // [N][C] f32
    const float* code_book = (const float*)d_in[1];  // [K][C] f32, K == C (square)
    (void)n_in; (void)out_size; (void)ws_size;

    // code_book is square: C = isqrt(in_sizes[1])
    int C = 1;
    while ((long long)(C + 1) * (long long)(C + 1) <= (long long)in_sizes[1]) ++C;
    const int K = C;
    const int N = in_sizes[0] / C;

    float* osum = (float*)d_ws;     // [N]
    float* csum = osum + N;         // [K]

    rowsum_kernel<<<N, 256, 0, stream>>>(out_feats, osum, C);
    rowsum_kernel<<<K, 256, 0, stream>>>(code_book, csum, C);

    dim3 grid((K + BN - 1) / BN, (N + BM - 1) / BM);
    hamming_gemm_kernel<<<grid, 256, 0, stream>>>(out_feats, code_book, osum, csum,
                                                  (float*)d_out, N, K, C);
}
